// RadarConv_14482629722249
// MI455X (gfx1250) — matrix operand order repaired and checked by reading, not compiled, not run on hardware
//
#include <hip/hip_runtime.h>
#include <hip/hip_bf16.h>
#include <math.h>

typedef __attribute__((ext_vector_type(16))) _Float16 v16h;
typedef __attribute__((ext_vector_type(8)))  float    v8f;

union Frag {
    uint4 u4[2];
    unsigned u[8];
    v16h h;
};

#define HH 80
#define WW 80
#define CC 256
#define HWP (HH*WW)          // 6400 pixels per image
#define NPIX (4*HWP)         // 25600 total pixels
#define KTILES 72            // 2304 / 32

__device__ __forceinline__ unsigned pack2h(float a, float b) {
    _Float16 ha = (_Float16)a, hb = (_Float16)b;
    unsigned short ua = __builtin_bit_cast(unsigned short, ha);
    unsigned short ub = __builtin_bit_cast(unsigned short, hb);
    return (unsigned)ua | ((unsigned)ub << 16);
}

__device__ __forceinline__ v8f wmma_f16(const Frag& a, const Frag& b, v8f c) {
    return __builtin_amdgcn_wmma_f32_16x16x32_f16(
        /*neg_a=*/false, a.h, /*neg_b=*/false, b.h,
        /*c_mod=*/(short)0, c, /*reuse_a=*/false, /*reuse_b=*/false);
}

// ---------------------------------------------------------------------------
// Kernel 1: 3x3 avg pool, NCHW -> NHWC (channels-last for coalesced gathers)
// ---------------------------------------------------------------------------
__global__ __launch_bounds__(256) void pool_nhwc(const float* __restrict__ x,
                                                 float* __restrict__ xp) {
    size_t idx = (size_t)blockIdx.x * 256 + threadIdx.x;   // lanes over w
    int w = (int)(idx % WW);
    size_t r = idx / WW;
    int h = (int)(r % HH); r /= HH;
    int c = (int)(r % CC);
    int b = (int)(r / CC);
    const float* src = x + (((size_t)b * CC + c) * HH) * WW;
    float s = 0.f;
#pragma unroll
    for (int dy = -1; dy <= 1; ++dy) {
#pragma unroll
        for (int dx = -1; dx <= 1; ++dx) {
            int y = h + dy, xx = w + dx;
            if ((unsigned)y < HH && (unsigned)xx < WW) s += src[(size_t)y * WW + xx];
        }
    }
    xp[(((size_t)b * HH + h) * WW + w) * CC + c] = s * (1.f / 9.f);
}

// ---------------------------------------------------------------------------
// Kernel 2: pack f32 weights into per-lane WMMA B-fragments (f16).
// B 32x16 order used here (contiguous k): col n = lane&15, K = 16*(lane>=16) + 2*v + j, v = VGPR 0..7.
// dst[(tileIdx*32 + lane)*8 + v] = packed pair.  tileIdx = ntile*72 + ktile.
// K ordering matches A: K = kk*256 + c;  w element = W[o][c][kk].
// ---------------------------------------------------------------------------
__global__ __launch_bounds__(256) void pack_wf16(const float* __restrict__ W0, int n0,
                                                 const float* __restrict__ W1, int n1,
                                                 unsigned* __restrict__ dst) {
    int t = threadIdx.x;
    int lane = t & 31;
    int v = t >> 5;                       // 0..7
    int tileIdx = blockIdx.x;
    int nt = tileIdx / KTILES;
    int kt = tileIdx % KTILES;
    int o = nt * 16 + (lane & 15);
    int K0 = kt * 32 + ((lane >> 4) << 4) + 2 * v;
    float f[2];
#pragma unroll
    for (int j = 0; j < 2; ++j) {
        int K = K0 + j;
        int kk = K >> 8;                  // K = kk*256 + c
        int c = K & 255;
        float val = 0.f;
        if (o < n0)           val = W0[((size_t)o * CC + c) * 9 + kk];
        else if (o < n0 + n1) val = W1[((size_t)(o - n0) * CC + c) * 9 + kk];
        f[j] = val;
    }
    dst[((size_t)tileIdx * 32 + lane) * 8 + v] = pack2h(f[0], f[1]);
}

// ---------------------------------------------------------------------------
// A-fragment gather from LDS chunk (rows padded to 264 halfs, bank-friendly).
// A 16x32 order used here (contiguous k): row m = lane&15; lanes<16 hold K 0-15, lanes>=16
// hold K 16-31; within a VGPR pair K = 16*(lane>=16) + 2v + j.
// ---------------------------------------------------------------------------
__device__ __forceinline__ void load_afrag(Frag& a, const _Float16* row, int kb) {
    a.u4[0] = *(const uint4*)(row + 2 * kb);
    a.u4[1] = *(const uint4*)(row + 2 * kb + 8);
}

__device__ __forceinline__ void load_bfrag(Frag& b, const unsigned* __restrict__ wp,
                                           int tileIdx, int lane) {
    const uint4* p = (const uint4*)(wp + (((size_t)tileIdx * 32 + lane) << 3));
    b.u4[0] = p[0];
    b.u4[1] = p[1];
}

// ---------------------------------------------------------------------------
// Kernel 3: offset+mask 3x3 conv via WMMA implicit GEMM.
// 64 pixels/block, 8 waves: wave = (subtile 0..3) x (ntile 0..1).
// Out: meta[p*32 + ch], ch 0..17 offsets (dy0,dx0,...), 18..26 mask (2*sigmoid)
// ---------------------------------------------------------------------------
__global__ __launch_bounds__(256) void offmask_conv(const float* __restrict__ xp,
                                                    const unsigned* __restrict__ wpack2,
                                                    const float* __restrict__ off_b,
                                                    const float* __restrict__ mod_b,
                                                    float* __restrict__ meta) {
    __shared__ __align__(16) _Float16 As[64][264];
    __shared__ float Ds[8][16][17];

    const int t = threadIdx.x;
    const int lane = t & 31;
    const int wid = t >> 5;
    const int p0 = blockIdx.x * 64;
    const int st = wid >> 1;        // pixel subtile
    const int nt = wid & 1;         // output-channel tile
    const int nloc = lane & 15;
    const int kb = (lane >> 4) << 3;
    const int mrow = st * 16 + nloc;

    v8f acc = {0.f,0.f,0.f,0.f,0.f,0.f,0.f,0.f};

    for (int kk = 0; kk < 9; ++kk) {
        int ki = kk / 3, kj = kk % 3;
        // build 64x256 f16 A-chunk (plain im2col, zero pad)
        int cpair = (t & 127) * 2;
        for (int mm = (t >> 7); mm < 64; mm += 2) {
            int p = p0 + mm;
            int b = p / HWP;
            int rem = p - b * HWP;
            int h = rem / WW, w = rem % WW;
            int y = h - 1 + ki, x = w - 1 + kj;
            float v0 = 0.f, v1 = 0.f;
            if ((unsigned)y < HH && (unsigned)x < WW) {
                const float2 av = *(const float2*)(xp + ((((size_t)b * HH + y) * WW + x) * CC + cpair));
                v0 = av.x; v1 = av.y;
            }
            *(unsigned*)&As[mm][cpair] = pack2h(v0, v1);
        }
        __syncthreads();
#pragma unroll
        for (int ks = 0; ks < 8; ++ks) {
            Frag a, b;
            load_afrag(a, &As[mrow][ks * 32], kb);
            load_bfrag(b, wpack2, nt * KTILES + kk * 8 + ks, lane);
            acc = wmma_f16(a, b, acc);
        }
        __syncthreads();
    }

    // epilogue: bias, sigmoid for mask channels, LDS transpose, coalesced store
    int o = nt * 16 + nloc;
    float bv = 0.f;
    bool isMask = false;
    if (o < 18) bv = off_b[o];
    else if (o < 27) { bv = mod_b[o - 18]; isMask = true; }
    const int mhalf = (lane >> 4) << 3;
#pragma unroll
    for (int r = 0; r < 8; ++r) {
        float v = acc[r] + bv;
        v = isMask ? (2.f / (1.f + __expf(-v))) : v;
        Ds[wid][r + mhalf][nloc] = v;
    }
    __builtin_amdgcn_wave_barrier();
#pragma unroll
    for (int rr = 0; rr < 8; ++rr) {
        int ol = rr * 2 + (lane >> 4);
        int px = nloc;
        int p = p0 + st * 16 + px;
        meta[(size_t)p * 32 + nt * 16 + ol] = Ds[wid][px][ol];
    }
}

// ---------------------------------------------------------------------------
// Kernel 4: main deformable conv.  32 pixels/block, 8 waves; each wave owns a
// 2x2 register tile (2 M-tiles x 2 N-tiles) so every A and B fragment feeds
// two v_wmma_f32_16x16x32_f16 ops (double arithmetic intensity, half the
// L2 B-fragment refetch).  Per tap kk: build the bilinear-sampled,
// mask-modulated 32x256 f16 A-chunk in LDS, then 8 K-subchunks x 4 WMMAs.
// ---------------------------------------------------------------------------
__global__ __launch_bounds__(256) void deform_main(const float* __restrict__ xp,
                                                   const float* __restrict__ meta,
                                                   const unsigned* __restrict__ wpack,
                                                   const float* __restrict__ bias,
                                                   float* __restrict__ out) {
    __shared__ __align__(16) _Float16 As[32][264];
    __shared__ float Dsa[8][16][17];
    __shared__ float Dsb[8][16][17];
    __shared__ int   sy0[288], sx0[288];
    __shared__ float swy[288], swx[288], smk[288];

    const int t = threadIdx.x;
    const int lane = t & 31;
    const int wid = t >> 5;
    const int p0 = blockIdx.x * 32;
    const int nloc = lane & 15;
    const int kb = (lane >> 4) << 3;
    const int nt0 = wid * 2, nt1 = wid * 2 + 1;

    // per-(pixel,tap) sampling metadata: 32 px * 9 taps = 288 points
    for (int q = t; q < 288; q += 256) {
        int m = q / 9, kk = q % 9;
        int p = p0 + m;
        int b = p / HWP;
        int rem = p - b * HWP;
        int h = rem / WW, w = rem % WW;
        float dy = meta[(size_t)p * 32 + 2 * kk];
        float dx = meta[(size_t)p * 32 + 2 * kk + 1];
        float mk = meta[(size_t)p * 32 + 18 + kk];
        float py = (float)(h - 1 + kk / 3) + dy;
        float px = (float)(w - 1 + kk % 3) + dx;
        float fy = floorf(py), fx = floorf(px);
        sy0[q] = (int)fy; sx0[q] = (int)fx;
        swy[q] = py - fy; swx[q] = px - fx;
        smk[q] = mk;
    }
    __syncthreads();

    v8f acc00 = {0.f,0.f,0.f,0.f,0.f,0.f,0.f,0.f};   // Mtile0 x Ntile0
    v8f acc01 = {0.f,0.f,0.f,0.f,0.f,0.f,0.f,0.f};   // Mtile0 x Ntile1
    v8f acc10 = {0.f,0.f,0.f,0.f,0.f,0.f,0.f,0.f};   // Mtile1 x Ntile0
    v8f acc11 = {0.f,0.f,0.f,0.f,0.f,0.f,0.f,0.f};   // Mtile1 x Ntile1

    for (int kk = 0; kk < 9; ++kk) {
        // build sampled A-chunk: each thread covers one channel pair, 16 pixels
        int cpair = (t & 127) * 2;
        for (int mm = (t >> 7); mm < 32; mm += 2) {
            int q = mm * 9 + kk;
            int p = p0 + mm;
            int b = p / HWP;
            int y0 = sy0[q], x0 = sx0[q];
            float wy = swy[q], wx = swx[q], mk = smk[q];
            float w00 = (1.f - wy) * (1.f - wx) * mk;
            float w01 = (1.f - wy) * wx * mk;
            float w10 = wy * (1.f - wx) * mk;
            float w11 = wy * wx * mk;
            float v0 = 0.f, v1 = 0.f;
            const float* base = xp + (size_t)b * HWP * CC + cpair;
#pragma unroll
            for (int tap = 0; tap < 4; ++tap) {
                int y = y0 + (tap >> 1);
                int x = x0 + (tap & 1);
                float wgt = (tap == 0) ? w00 : (tap == 1) ? w01 : (tap == 2) ? w10 : w11;
                if ((unsigned)y < HH && (unsigned)x < WW) {
                    const float2 av = *(const float2*)(base + ((size_t)y * WW + x) * CC);
                    v0 += wgt * av.x;
                    v1 += wgt * av.y;
                }
            }
            *(unsigned*)&As[mm][cpair] = pack2h(v0, v1);
        }
        __syncthreads();
#pragma unroll
        for (int ks = 0; ks < 8; ++ks) {
            Frag a0, a1, b0, b1;
            load_afrag(a0, &As[nloc][ks * 32], kb);
            load_afrag(a1, &As[16 + nloc][ks * 32], kb);
            load_bfrag(b0, wpack, nt0 * KTILES + kk * 8 + ks, lane);
            load_bfrag(b1, wpack, nt1 * KTILES + kk * 8 + ks, lane);
            acc00 = wmma_f16(a0, b0, acc00);
            acc01 = wmma_f16(a0, b1, acc01);
            acc10 = wmma_f16(a1, b0, acc10);
            acc11 = wmma_f16(a1, b1, acc11);
        }
        __syncthreads();
    }

    // epilogue: bias + LDS transpose -> 64B-coalesced NCHW stores
    const float bv0 = bias[nt0 * 16 + nloc];
    const float bv1 = bias[nt1 * 16 + nloc];
    const int mhalf = (lane >> 4) << 3;
    const int b = p0 / HWP;                 // 6400 % 32 == 0: tile in one batch
    float* outbase = out + (size_t)b * CC * HWP + (p0 - b * HWP);

#pragma unroll
    for (int mt = 0; mt < 2; ++mt) {
        const v8f& a0 = (mt == 0) ? acc00 : acc10;
        const v8f& a1 = (mt == 0) ? acc01 : acc11;
#pragma unroll
        for (int r = 0; r < 8; ++r) {
            Dsa[wid][r + mhalf][nloc] = a0[r] + bv0;
            Dsb[wid][r + mhalf][nloc] = a1[r] + bv1;
        }
        __builtin_amdgcn_wave_barrier();
        float* outb = outbase + mt * 16;    // pixel offset of this M-tile
#pragma unroll
        for (int rr = 0; rr < 8; ++rr) {
            int ol = rr * 2 + (lane >> 4);
            int px = nloc;
            outb[(size_t)(nt0 * 16 + ol) * HWP + px] = Dsa[wid][px][ol];
            outb[(size_t)(nt1 * 16 + ol) * HWP + px] = Dsb[wid][px][ol];
        }
        __builtin_amdgcn_wave_barrier();    // before Ds reuse by next M-tile
    }
}

// ---------------------------------------------------------------------------
extern "C" void kernel_launch(void* const* d_in, const int* in_sizes, int n_in,
                              void* d_out, int out_size, void* d_ws, size_t ws_size,
                              hipStream_t stream) {
    (void)in_sizes; (void)n_in; (void)out_size; (void)ws_size;
    const float* x     = (const float*)d_in[0];
    const float* off_w = (const float*)d_in[1];
    const float* off_b = (const float*)d_in[2];
    const float* mod_w = (const float*)d_in[3];
    const float* mod_b = (const float*)d_in[4];
    const float* w     = (const float*)d_in[5];
    const float* bias  = (const float*)d_in[6];
    float* out = (float*)d_out;

    // workspace layout (all offsets 16B aligned)
    char* ws = (char*)d_ws;
    float*    xp        = (float*)ws;                           // 26,214,400 B
    float*    meta      = (float*)(ws + 26214400);              //  3,276,800 B
    unsigned* wpackMain = (unsigned*)(ws + 26214400 + 3276800); //  1,179,648 B
    unsigned* wpack2    = (unsigned*)(ws + 26214400 + 3276800 + 1179648); // 147,456 B

    // 1) avg-pool to NHWC
    pool_nhwc<<<(NPIX * CC) / 256, 256, 0, stream>>>(x, xp);
    // 2) pre-pack weights into WMMA B-fragments (f16)
    pack_wf16<<<16 * KTILES, 256, 0, stream>>>(w, 256, nullptr, 0, wpackMain);
    pack_wf16<<<2 * KTILES, 256, 0, stream>>>(off_w, 18, mod_w, 9, wpack2);
    // 3) offset + mask conv (WMMA), fused 2*sigmoid
    offmask_conv<<<NPIX / 64, 256, 0, stream>>>(xp, wpack2, off_b, mod_b, meta);
    // 4) main deformable conv (WMMA)
    deform_main<<<NPIX / 32, 256, 0, stream>>>(xp, meta, wpackMain, bias, out);
}
